// Model_80977313399128
// MI455X (gfx1250) — matrix operand order repaired and checked by reading, not compiled, not run on hardware
//
#include <hip/hip_runtime.h>

typedef __attribute__((ext_vector_type(16))) _Float16 v16h;
typedef __attribute__((ext_vector_type(8)))  _Float16 v8h;
typedef __attribute__((ext_vector_type(8)))  float    v8f;

#define FT_IN 128
#define NHID  64
#define W_ELEMS (NHID * FT_IN)   // 8192 f16 = 16 KB

// ---------------------------------------------------------------------------
// Prep: convert W_enc [64x128] and W1 [64x128] from f32 to f16 in workspace.
// ---------------------------------------------------------------------------
__global__ void cvt_weights_kernel(const float* __restrict__ Wenc,
                                   const float* __restrict__ W1,
                                   _Float16* __restrict__ WencH,
                                   _Float16* __restrict__ W1H) {
  int i = blockIdx.x * blockDim.x + threadIdx.x;
  if (i < W_ELEMS) {
    WencH[i] = (_Float16)Wenc[i];
    W1H[i]   = (_Float16)W1[i];
  }
}

// Stage the 16 KB f16 weight matrix into LDS (cooperative, whole block).
__device__ __forceinline__ void stage_weights(_Float16* __restrict__ sW,
                                              const _Float16* __restrict__ WH) {
  for (int i = threadIdx.x; i < W_ELEMS / 8; i += blockDim.x)
    ((v8h*)sW)[i] = ((const v8h*)WH)[i];
  __syncthreads();
}

// ---------------------------------------------------------------------------
// Encoder (full tiles only): z = seq @ W_enc^T + b_enc -> f16 [n x 64].
// One wave per 16-node tile; 16 WMMAs with 4 interleaved accumulator chains.
// ---------------------------------------------------------------------------
__global__ void __launch_bounds__(256)
encoder_wmma_kernel(const float* __restrict__ seq,
                    const _Float16* __restrict__ WencH,
                    const float* __restrict__ b_enc,
                    _Float16* __restrict__ zH,
                    int n_tiles) {
  __shared__ _Float16 sW[W_ELEMS];
  stage_weights(sW, WencH);

  const int lane = threadIdx.x & 31;
  const int wave = threadIdx.x >> 5;
  const int wpb  = blockDim.x >> 5;
  const int half = lane >> 4;        // 0: lanes 0-15, 1: lanes 16-31
  const int idx  = lane & 15;
  const int kbA  = half * 16;        // A-fragment K sub-base
  const int kbB  = half * 16;        // B-fragment K sub-base

  float bias[4];
#pragma unroll
  for (int t = 0; t < 4; ++t) bias[t] = b_enc[t * 16 + idx];

  for (int tile = blockIdx.x * wpb + wave; tile < n_tiles;
       tile += (int)gridDim.x * wpb) {
    const float* srow = seq + (long)(tile * 16 + idx) * FT_IN;

    // A fragments: load f32, convert, pack contiguous k = 16*half + s (same order as B).
    v16h af[4];
#pragma unroll
    for (int c = 0; c < 4; ++c) {
      v8f flo = *(const v8f*)(srow + c * 32 + kbA);
      v8f fhi = *(const v8f*)(srow + c * 32 + 8 + kbA);
      v8h lo, hi;
#pragma unroll
      for (int i = 0; i < 8; ++i) { lo[i] = (_Float16)flo[i]; hi[i] = (_Float16)fhi[i]; }
      af[c] = __builtin_shufflevector(lo, hi, 0, 1, 2, 3, 4, 5, 6, 7,
                                      8, 9, 10, 11, 12, 13, 14, 15);
    }

    // 16 WMMAs, 4 independent accumulator chains interleaved.
    v8f acc[4];
#pragma unroll
    for (int t = 0; t < 4; ++t) acc[t] = (v8f){};
#pragma unroll
    for (int c = 0; c < 4; ++c) {
#pragma unroll
      for (int t = 0; t < 4; ++t) {
        v16h bf = *(const v16h*)(sW + (t * 16 + idx) * FT_IN + c * 32 + kbB);
        acc[t] = __builtin_amdgcn_wmma_f32_16x16x32_f16(
            false, af[c], false, bf, (short)0, acc[t], false, false);
      }
    }

    // D layout: VGPR v -> row (v + 8*half), col (idx + 16*t). Unpredicated.
#pragma unroll
    for (int t = 0; t < 4; ++t) {
      _Float16* zp = zH + (long)(tile * 16 + 8 * half) * NHID + t * 16 + idx;
#pragma unroll
      for (int v = 0; v < 8; ++v)
        zp[v * NHID] = (_Float16)(acc[t][v] + bias[t]);
    }
  }
}

// Scalar tail for nodes beyond the last full tile (not launched when n%16==0).
__global__ void encoder_tail_kernel(const float* __restrict__ seq,
                                    const _Float16* __restrict__ WencH,
                                    const float* __restrict__ b_enc,
                                    _Float16* __restrict__ zH,
                                    int n_nodes, int start) {
  int n = start + blockIdx.x * blockDim.x + threadIdx.x;
  if (n >= n_nodes) return;
  for (int h = 0; h < NHID; ++h) {
    float a = 0.0f;
    for (int k = 0; k < FT_IN; ++k)
      a += (float)(_Float16)seq[(long)n * FT_IN + k] * (float)WencH[h * FT_IN + k];
    zH[(long)n * NHID + h] = (_Float16)(a + b_enc[h]);
  }
}

// ---------------------------------------------------------------------------
// Decoder (full tiles only): a=[z[row],z[col]] f16; h=relu(a@W1^T+b1) f32;
// out = h.W2 + b2. One wave per 16-edge tile.
// ---------------------------------------------------------------------------
__global__ void __launch_bounds__(256)
decoder_wmma_kernel(const _Float16* __restrict__ zH,
                    const _Float16* __restrict__ W1H,
                    const float* __restrict__ b1,
                    const float* __restrict__ W2,
                    const float* __restrict__ b2,
                    const int* __restrict__ rowi,
                    const int* __restrict__ coli,
                    float* __restrict__ out,
                    int n_tiles) {
  __shared__ _Float16 sW[W_ELEMS];
  stage_weights(sW, W1H);

  const int lane = threadIdx.x & 31;
  const int wave = threadIdx.x >> 5;
  const int wpb  = blockDim.x >> 5;
  const int half = lane >> 4;
  const int idx  = lane & 15;
  const int kbA  = half * 16;
  const int kbB  = half * 16;

  float bias1[4], w2v[4];
#pragma unroll
  for (int t = 0; t < 4; ++t) {
    bias1[t] = b1[t * 16 + idx];
    w2v[t]   = W2[t * 16 + idx];
  }
  const float b2s = b2[0];

  for (int tile = blockIdx.x * wpb + wave; tile < n_tiles;
       tile += (int)gridDim.x * wpb) {
    int e = tile * 16 + idx;
    const _Float16* zr = zH + (long)rowi[e] * NHID;
    const _Float16* zc = zH + (long)coli[e] * NHID;

    // A fragments: chunks 0,1 from z[row], chunks 2,3 from z[col];
    // two 16 B contiguous loads each from the L2-resident z table.
    v16h af[4];
#pragma unroll
    for (int c = 0; c < 4; ++c) {
      const _Float16* src = (c < 2) ? (zr + c * 32) : (zc + (c - 2) * 32);
      v8h lo = *(const v8h*)(src + kbA);
      v8h hi = *(const v8h*)(src + 8 + kbA);
      af[c] = __builtin_shufflevector(lo, hi, 0, 1, 2, 3, 4, 5, 6, 7,
                                      8, 9, 10, 11, 12, 13, 14, 15);
    }

    // 16 WMMAs, 4 interleaved accumulator chains; VALU reads accs once.
    v8f acc[4];
#pragma unroll
    for (int t = 0; t < 4; ++t) acc[t] = (v8f){};
#pragma unroll
    for (int c = 0; c < 4; ++c) {
#pragma unroll
      for (int t = 0; t < 4; ++t) {
        v16h bf = *(const v16h*)(sW + (t * 16 + idx) * FT_IN + c * 32 + kbB);
        acc[t] = __builtin_amdgcn_wmma_f32_16x16x32_f16(
            false, af[c], false, bf, (short)0, acc[t], false, false);
      }
    }

    // bias + relu + lin2 partials in registers.
    float p[8];
#pragma unroll
    for (int v = 0; v < 8; ++v) p[v] = 0.0f;
#pragma unroll
    for (int t = 0; t < 4; ++t) {
#pragma unroll
      for (int v = 0; v < 8; ++v) {
        float h = acc[t][v] + bias1[t];
        h = h > 0.0f ? h : 0.0f;
        p[v] += w2v[t] * h;
      }
    }

    // Sum over hidden dim: butterfly across the 16 lanes of each half.
#pragma unroll
    for (int m = 1; m < 16; m <<= 1) {
#pragma unroll
      for (int v = 0; v < 8; ++v) p[v] += __shfl_xor(p[v], m, 16);
    }

    // All lanes store (lane pairs duplicate identical values): no exec mask.
    out[tile * 16 + 8 * half + (idx & 7)] = p[idx & 7] + b2s;
  }
}

// Scalar tail for edges beyond the last full tile (not launched when E%16==0).
__global__ void decoder_tail_kernel(const _Float16* __restrict__ zH,
                                    const _Float16* __restrict__ W1H,
                                    const float* __restrict__ b1,
                                    const float* __restrict__ W2,
                                    const float* __restrict__ b2,
                                    const int* __restrict__ rowi,
                                    const int* __restrict__ coli,
                                    float* __restrict__ out,
                                    int n_edges, int start) {
  int e = start + blockIdx.x * blockDim.x + threadIdx.x;
  if (e >= n_edges) return;
  const _Float16* zr = zH + (long)rowi[e] * NHID;
  const _Float16* zc = zH + (long)coli[e] * NHID;
  float o = b2[0];
  for (int h = 0; h < NHID; ++h) {
    float a = 0.0f;
    for (int k = 0; k < NHID; ++k) {
      a += (float)zr[k] * (float)W1H[h * FT_IN + k];
      a += (float)zc[k] * (float)W1H[h * FT_IN + NHID + k];
    }
    a += b1[h];
    o += W2[h] * (a > 0.0f ? a : 0.0f);
  }
  out[e] = o;
}

// ---------------------------------------------------------------------------
extern "C" void kernel_launch(void* const* d_in, const int* in_sizes, int n_in,
                              void* d_out, int out_size, void* d_ws, size_t ws_size,
                              hipStream_t stream) {
  const float* seq   = (const float*)d_in[0];
  const float* W_enc = (const float*)d_in[1];
  const float* b_enc = (const float*)d_in[2];
  const float* W1    = (const float*)d_in[3];
  const float* b1    = (const float*)d_in[4];
  const float* W2    = (const float*)d_in[5];
  const float* b2    = (const float*)d_in[6];
  const int*   row   = (const int*)d_in[7];
  const int*   col   = (const int*)d_in[8];
  float* out = (float*)d_out;

  const int n_nodes = in_sizes[0] / FT_IN;
  const int n_edges = in_sizes[7];

  char* ws = (char*)d_ws;
  _Float16* WencH = (_Float16*)ws;                 // 16 KB
  _Float16* W1H   = (_Float16*)(ws + 16384);       // 16 KB
  _Float16* zH    = (_Float16*)(ws + 32768);       // n_nodes*64*2 (~12.8 MB)

  cvt_weights_kernel<<<(W_ELEMS + 255) / 256, 256, 0, stream>>>(
      W_enc, W1, WencH, W1H);

  const int tiles_n = n_nodes >> 4;                // full tiles only
  if (tiles_n > 0) {
    const int blocks1 = (tiles_n + 7) / 8;         // 8 waves/block, 1 tile/wave
    encoder_wmma_kernel<<<blocks1, 256, 0, stream>>>(
        seq, WencH, b_enc, zH, tiles_n);
  }
  if (n_nodes & 15)                                // remainder (never for 100000)
    encoder_tail_kernel<<<1, 16, 0, stream>>>(
        seq, WencH, b_enc, zH, n_nodes, tiles_n * 16);

  const int tiles_e = n_edges >> 4;
  if (tiles_e > 0) {
    const int blocks2 = (tiles_e + 7) / 8;
    decoder_wmma_kernel<<<blocks2, 256, 0, stream>>>(
        zH, W1H, b1, W2, b2, row, col, out, tiles_e);
  }
  if (n_edges & 15)                                // remainder (never for 1e6)
    decoder_tail_kernel<<<1, 16, 0, stream>>>(
        zH, W1H, b1, W2, b2, row, col, out, n_edges, tiles_e * 16);
}
